// RowAttentionBlock_73289321939576
// MI455X (gfx1250) — matrix operand order repaired and checked by reading, not compiled, not run on hardware
//
#include <hip/hip_runtime.h>
#include <math.h>

// ---------------------------------------------------------------------------
// RowAttentionBlock for MI455X (gfx1250), B=1, I=J=1536, CN=512, CP=64, CH=64, H=8
// Memory-bound on pair_embed (603 MB fp32). All GEMM-ish work on v_wmma bf16.
// ---------------------------------------------------------------------------

#define IDIM  1536
#define JDIM  1536
#define CND   512
#define CPD   64
#define CHD   64
#define NH    8
#define HCD   512          // NH * CHD
#define EPSLN 1e-5f
#define BIGINF 1e9f

typedef __attribute__((ext_vector_type(16))) __bf16 v16bf;
typedef __attribute__((ext_vector_type(8)))  float  v8f;

__device__ __forceinline__ v8f wmma_bf16(v16bf a, v16bf b, v8f c) {
  // D = A(16x32 bf16) x B(32x16 bf16) + C(16x16 f32)
  return __builtin_amdgcn_wmma_f32_16x16x32_bf16(false, a, false, b, (short)0, c,
                                                 false, false);
}

// A-operand (16-bit A 16x32, ISA 7.12.2): lane (hf, m) holds row m with
//   elems 0..7  = row[kb + hf*8 + t]
//   elems 8..15 = row[kb + 16 + hf*8 + t]
__device__ __forceinline__ v16bf make_a(const float* __restrict__ row, int kb, int hf) {
  v16bf a;
  const float* p0 = row + kb + hf * 8;
  const float* p1 = p0 + 16;
#pragma unroll
  for (int t = 0; t < 8; ++t) {
    a[t]     = (__bf16)p0[t];
    a[8 + t] = (__bf16)p1[t];
  }
  return a;
}

// B-operand (16-bit B 32x16): lane (hf, n) holds column n with
//   elems 0..7  = col[kb + hf*8 + t]
//   elems 8..15 = col[kb + 16 + hf*8 + t]  (col = contiguous K run).
__device__ __forceinline__ v16bf make_b(const float* __restrict__ col, int kb, int hf) {
  v16bf b;
  const float* p = col + kb + hf * 8;
#pragma unroll
  for (int t = 0; t < 8; ++t) {
    b[t]     = (__bf16)p[t];
    b[8 + t] = (__bf16)p[16 + t];
  }
  return b;
}

// ---------------------------------------------------------------------------
// 1) LayerNorm of node embeddings: one block (256 thr) per 512-wide row.
//    torch-style: var ddof=1, divide by (std + eps).
// ---------------------------------------------------------------------------
__global__ void ln_node_kernel(const float* __restrict__ x, const float* __restrict__ ga,
                               const float* __restrict__ gb, float* __restrict__ y) {
  int row = blockIdx.x;
  int t = threadIdx.x;
  const float* xr = x + (size_t)row * CND;
  float* yr = y + (size_t)row * CND;
  float v0 = xr[t], v1 = xr[t + 256];
  __shared__ float red[256];
  red[t] = v0 + v1;
  __syncthreads();
  for (int off = 128; off > 0; off >>= 1) {
    if (t < off) red[t] += red[t + off];
    __syncthreads();
  }
  float mean = red[0] * (1.0f / CND);
  __syncthreads();
  float d0 = v0 - mean, d1 = v1 - mean;
  red[t] = d0 * d0 + d1 * d1;
  __syncthreads();
  for (int off = 128; off > 0; off >>= 1) {
    if (t < off) red[t] += red[t + off];
    __syncthreads();
  }
  float inv = 1.0f / (sqrtf(red[0] * (1.0f / (CND - 1))) + EPSLN);
  yr[t]       = ga[t] * d0 * inv + gb[t];
  yr[t + 256] = ga[t + 256] * d1 * inv + gb[t + 256];
}

// ---------------------------------------------------------------------------
// 2) q/k/v/g projections. One wave per (matrix, 16x32 output tile pair), K=512.
//    Two N-tiles share the A operand: halves A traffic and provides an
//    independent WMMA to cover the WMMA->VALU WAR hazard window.
//    q scaled 1/sqrt(64); g = sigmoid(.+bg); v stored transposed [HCD][J].
// ---------------------------------------------------------------------------
__global__ void proj_qkvg_kernel(const float* __restrict__ ni, const float* __restrict__ nj,
                                 const float* __restrict__ Wq, const float* __restrict__ Wk,
                                 const float* __restrict__ Wv, const float* __restrict__ Wg,
                                 const float* __restrict__ bg,
                                 float* __restrict__ q, float* __restrict__ k,
                                 float* __restrict__ vT, float* __restrict__ g) {
  int wid  = blockIdx.x * 8 + (threadIdx.x >> 5);   // 0..6143
  int lane = threadIdx.x & 31;
  int hf = lane >> 4, ln15 = lane & 15;
  int mat  = wid / 1536;          // 0:q 1:k 2:v 3:g
  int tile = wid % 1536;
  int it = tile >> 4;             // row tile 0..95
  int np = tile & 15;             // col-tile pair 0..15 -> cols np*32 .. np*32+31

  const float* src = (mat == 0 || mat == 3) ? ni : nj;
  const float* W   = (mat == 0) ? Wq : (mat == 1) ? Wk : (mat == 2) ? Wv : Wg;

  const float* arow  = src + (size_t)(it * 16 + ln15) * CND;        // A row m = ln15
  const float* b0col = W   + (size_t)(np * 32 + ln15) * CND;        // B col n (tile 0)
  const float* b1col = W   + (size_t)(np * 32 + 16 + ln15) * CND;   // B col n (tile 1)

  v8f acc0 = {}, acc1 = {};
#pragma unroll 4
  for (int kk = 0; kk < CND; kk += 32) {
    v16bf a  = make_a(arow, kk, hf);
    v16bf b0 = make_b(b0col, kk, hf);
    v16bf b1 = make_b(b1col, kk, hf);
    acc0 = wmma_bf16(a, b0, acc0);
    acc1 = wmma_bf16(a, b1, acc1);
  }

  int c0 = np * 32 + ln15, c1 = c0 + 16;
  if (mat == 0) {
#pragma unroll
    for (int r = 0; r < 8; ++r) {
      size_t rowo = (size_t)(it * 16 + r + 8 * hf) * HCD;
      q[rowo + c0] = acc0[r] * 0.125f;
      q[rowo + c1] = acc1[r] * 0.125f;
    }
  } else if (mat == 1) {
#pragma unroll
    for (int r = 0; r < 8; ++r) {
      size_t rowo = (size_t)(it * 16 + r + 8 * hf) * HCD;
      k[rowo + c0] = acc0[r];
      k[rowo + c1] = acc1[r];
    }
  } else if (mat == 2) {
#pragma unroll
    for (int r = 0; r < 8; ++r) {
      int jj = it * 16 + r + 8 * hf;
      vT[(size_t)c0 * JDIM + jj] = acc0[r];
      vT[(size_t)c1 * JDIM + jj] = acc1[r];
    }
  } else {
    float bb0 = bg[c0], bb1 = bg[c1];
#pragma unroll
    for (int r = 0; r < 8; ++r) {
      size_t rowo = (size_t)(it * 16 + r + 8 * hf) * HCD;
      float z0 = acc0[r] + bb0, z1 = acc1[r] + bb1;
      g[rowo + c0] = 1.0f / (1.0f + __expf(-z0));
      g[rowo + c1] = 1.0f / (1.0f + __expf(-z1));
    }
  }
}

// ---------------------------------------------------------------------------
// 3) Pair bias: stream pair_embed once. One wave per (j, 16-wide i-tile):
//    LN over CP=64 in registers, 2x WMMA vs W_comb^T, fuse halves + b_comb +
//    mask bias, store transposed pb[h][j][i] (coalesced over i).
// ---------------------------------------------------------------------------
__global__ void pair_bias_kernel(const float* __restrict__ pe, const float* __restrict__ pm,
                                 const float* __restrict__ lpa, const float* __restrict__ lpb,
                                 const float* __restrict__ Wc, const float* __restrict__ bc,
                                 const float* __restrict__ bfuse,
                                 float* __restrict__ pb) {
  int warp = threadIdx.x >> 5;
  int lane = threadIdx.x & 31;
  int hf = lane >> 4, ln15 = lane & 15;
  int wid = blockIdx.x * 8 + warp;
  int itile = wid % 96;
  int j     = wid / 96;           // 0..1535
  int i0 = itile * 16;

  // B operands from W_comb [16, 64]: column n of B = row n of W_comb.
  const float* wrow = Wc + ln15 * CPD;
  v16bf b0 = make_b(wrow, 0, hf);
  v16bf b1 = make_b(wrow, 32, hf);

  // Load pe row (i0+ln15, j, :) once; accumulate LN stats; keep the 8 float4
  // chunks this lane contributes to the A operands (A layout, K chunks by hf).
  const float* row = pe + ((size_t)(i0 + ln15) * JDIM + j) * CPD;
  float4 kept[8];
  float s = 0.f, s2 = 0.f;
#pragma unroll
  for (int qq = 0; qq < 16; ++qq) {
    float4 f = ((const float4*)row)[qq];
    s  += f.x + f.y + f.z + f.w;
    s2 += f.x * f.x + f.y * f.y + f.z * f.z + f.w * f.w;
    if (((qq >> 1) & 1) == hf) kept[(qq >> 2) * 2 + (qq & 1)] = f;
  }
  float mean = s * (1.f / 64.f);
  float var = (s2 - 64.f * mean * mean) * (1.f / 63.f);
  var = var < 0.f ? 0.f : var;
  float inv = 1.f / (sqrtf(var) + EPSLN);

  // kept[c] covers channels: ch = hf*8 + (c>>1)*16 + (c&1)*4 .. +3
  v16bf a0, a1;
#pragma unroll
  for (int c = 0; c < 8; ++c) {
    float4 f = kept[c];
    int ch = hf * 8 + (c >> 1) * 16 + (c & 1) * 4;
    float e0 = lpa[ch + 0] * (f.x - mean) * inv + lpb[ch + 0];
    float e1 = lpa[ch + 1] * (f.y - mean) * inv + lpb[ch + 1];
    float e2 = lpa[ch + 2] * (f.z - mean) * inv + lpb[ch + 2];
    float e3 = lpa[ch + 3] * (f.w - mean) * inv + lpb[ch + 3];
    int eb = (c & 3) * 4;
    if (c < 4) {
      a0[eb + 0] = (__bf16)e0; a0[eb + 1] = (__bf16)e1;
      a0[eb + 2] = (__bf16)e2; a0[eb + 3] = (__bf16)e3;
    } else {
      a1[eb + 0] = (__bf16)e0; a1[eb + 1] = (__bf16)e1;
      a1[eb + 2] = (__bf16)e2; a1[eb + 3] = (__bf16)e3;
    }
  }

  v8f acc = {};
  acc = wmma_bf16(a0, b0, acc);
  acc = wmma_bf16(a1, b1, acc);

  // Fuse halves across lanes via LDS: C[m][n], m = i offset, n = comb index.
  __shared__ float Ct[8][256];
  float* C = Ct[warp];
#pragma unroll
  for (int r = 0; r < 8; ++r) C[(r + 8 * hf) * 16 + ln15] = acc[r];
  __syncthreads();
#pragma unroll
  for (int t = 0; t < 4; ++t) {
    int idx = t * 32 + lane;
    int h  = idx >> 4;
    int mm = idx & 15;
    float fused = (C[mm * 16 + h] + bc[h]) +
                  bfuse[h] * (C[mm * 16 + 8 + h] + bc[8 + h]);
    float mb = BIGINF * (pm[(size_t)(i0 + mm) * JDIM + j] - 1.0f);
    pb[((size_t)h * JDIM + j) * IDIM + i0 + mm] = fused + mb;
  }
}

// ---------------------------------------------------------------------------
// 4) Flash attention: one wave per (head, 16-row i-tile). Online softmax over
//    48 chunks of 32 j's. Scores/probs bounce through wave-private LDS to
//    convert C-layout -> A-layout.
// ---------------------------------------------------------------------------
#define ATT_WAVES 2
__global__ void attention_kernel(const float* __restrict__ q, const float* __restrict__ k,
                                 const float* __restrict__ vT, const float* __restrict__ pb,
                                 const float* __restrict__ g, float* __restrict__ o) {
  int warp = threadIdx.x >> 5;
  int lane = threadIdx.x & 31;
  int hf = lane >> 4, ln15 = lane & 15;
  int wid = blockIdx.x * ATT_WAVES + warp;   // 0..767
  int h = wid / 96;
  int i0 = (wid % 96) * 16;

  __shared__ float  S[ATT_WAVES][16 * 32];
  __shared__ __bf16 P[ATT_WAVES][16 * 32];
  __shared__ float  rmax[ATT_WAVES][16], rsum[ATT_WAVES][16], rscale[ATT_WAVES][16];
  float*  Sw = S[warp];
  __bf16* Pw = P[warp];

  const float* qrow = q + (size_t)(i0 + ln15) * HCD + h * CHD;
  v16bf aq0 = make_a(qrow, 0, hf);
  v16bf aq1 = make_a(qrow, 32, hf);

  v8f acc[4] = {};
  if (lane < 16) { rmax[warp][lane] = -1e30f; rsum[warp][lane] = 0.f; }
  __syncthreads();

  for (int jb = 0; jb < JDIM; jb += 32) {
    // prefetch next chunk of K and V^T streams
    if (jb + 32 < JDIM) {
      __builtin_prefetch(k + (size_t)(jb + 32 + ln15) * HCD + h * CHD, 0, 0);
      __builtin_prefetch(vT + (size_t)(h * CHD + ln15) * JDIM + jb + 32, 0, 0);
    }
    // ---- scores = qK^T + pair_bias(+mask), two 16-col subtiles ----
#pragma unroll
    for (int sub = 0; sub < 2; ++sub) {
      int j0 = jb + sub * 16;
      const float* krow = k + (size_t)(j0 + ln15) * HCD + h * CHD;
      v8f c = {};
      c = wmma_bf16(aq0, make_b(krow, 0, hf), c);
      c = wmma_bf16(aq1, make_b(krow, 32, hf), c);
      const float* pbcol = pb + ((size_t)h * JDIM + j0 + ln15) * IDIM + i0 + 8 * hf;
#pragma unroll
      for (int r = 0; r < 8; ++r)
        Sw[(r + 8 * hf) * 32 + sub * 16 + ln15] = c[r] + pbcol[r];
    }
    __syncthreads();
    // ---- online softmax: lanes 0..15 each own one row of 32 scores ----
    if (lane < 16) {
      float* srow = Sw + lane * 32;
      float mold = rmax[warp][lane];
      float mnew = mold;
#pragma unroll
      for (int t = 0; t < 32; ++t) mnew = fmaxf(mnew, srow[t]);
      float sc  = __expf(mold - mnew);
      float sum = rsum[warp][lane] * sc;
      __bf16* prow = Pw + lane * 32;
#pragma unroll
      for (int t = 0; t < 32; ++t) {
        float p = __expf(srow[t] - mnew);
        sum += p;
        prow[t] = (__bf16)p;
      }
      rmax[warp][lane] = mnew;
      rsum[warp][lane] = sum;
      rscale[warp][lane] = sc;
    }
    __syncthreads();
    // ---- rescale accumulators by per-row exp(mold-mnew) ----
    float scr[8];
#pragma unroll
    for (int r = 0; r < 8; ++r) scr[r] = rscale[warp][r + 8 * hf];
#pragma unroll
    for (int nc = 0; nc < 4; ++nc)
#pragma unroll
      for (int r = 0; r < 8; ++r) acc[nc][r] *= scr[r];
    // ---- probs (A layout) and PV accumulation (4 independent WMMAs) ----
    v16bf pa;
    const __bf16* pm0 = Pw + ln15 * 32 + hf * 8;
#pragma unroll
    for (int t = 0; t < 8; ++t) { pa[t] = pm0[t]; pa[8 + t] = pm0[16 + t]; }
#pragma unroll
    for (int nc = 0; nc < 4; ++nc) {
      const float* vrow = vT + (size_t)(h * CHD + nc * 16 + ln15) * JDIM + jb;
      acc[nc] = wmma_bf16(pa, make_b(vrow, 0, hf), acc[nc]);
    }
  }

  // ---- normalize, gate, store pre-projection output [I][HCD] ----
  float sums[8];
#pragma unroll
  for (int r = 0; r < 8; ++r) sums[r] = rsum[warp][r + 8 * hf];
#pragma unroll
  for (int nc = 0; nc < 4; ++nc)
#pragma unroll
    for (int r = 0; r < 8; ++r) {
      int i = i0 + r + 8 * hf;
      int d = h * CHD + nc * 16 + ln15;
      float gate = g[(size_t)i * HCD + d];
      o[(size_t)i * HCD + d] = acc[nc][r] / sums[r] * gate;
    }
}

// ---------------------------------------------------------------------------
// 5) Output projection: out = o @ Wo^T + bo. One wave per 16x32 tile pair,
//    K=512, shared A operand across the two N-tiles.
// ---------------------------------------------------------------------------
__global__ void out_proj_kernel(const float* __restrict__ o, const float* __restrict__ Wo,
                                const float* __restrict__ bo, float* __restrict__ out) {
  int wid  = blockIdx.x * 8 + (threadIdx.x >> 5);   // 0..1535
  int lane = threadIdx.x & 31;
  int hf = lane >> 4, ln15 = lane & 15;
  int it = wid >> 4;   // 0..95
  int np = wid & 15;   // 0..15 -> cols np*32 .. np*32+31
  const float* arow  = o  + (size_t)(it * 16 + ln15) * HCD;
  const float* b0col = Wo + (size_t)(np * 32 + ln15) * HCD;
  const float* b1col = Wo + (size_t)(np * 32 + 16 + ln15) * HCD;
  v8f acc0 = {}, acc1 = {};
#pragma unroll 4
  for (int kk = 0; kk < HCD; kk += 32) {
    v16bf a = make_a(arow, kk, hf);
    acc0 = wmma_bf16(a, make_b(b0col, kk, hf), acc0);
    acc1 = wmma_bf16(a, make_b(b1col, kk, hf), acc1);
  }
  int c0 = np * 32 + ln15, c1 = c0 + 16;
  float bb0 = bo[c0], bb1 = bo[c1];
#pragma unroll
  for (int r = 0; r < 8; ++r) {
    size_t rowo = (size_t)(it * 16 + r + 8 * hf) * CND;
    out[rowo + c0] = acc0[r] + bb0;
    out[rowo + c1] = acc1[r] + bb1;
  }
}

// ---------------------------------------------------------------------------
extern "C" void kernel_launch(void* const* d_in, const int* in_sizes, int n_in,
                              void* d_out, int out_size, void* d_ws, size_t ws_size,
                              hipStream_t stream) {
  (void)in_sizes; (void)n_in; (void)out_size; (void)ws_size;
  const float* node_i = (const float*)d_in[0];
  const float* node_j = (const float*)d_in[1];
  const float* pe     = (const float*)d_in[2];
  const float* pm     = (const float*)d_in[3];
  /* d_in[4] node_mask_i is unused by the reference */
  const float* ln_i_a = (const float*)d_in[5];
  const float* ln_i_b = (const float*)d_in[6];
  const float* ln_j_a = (const float*)d_in[7];
  const float* ln_j_b = (const float*)d_in[8];
  const float* ln_p_a = (const float*)d_in[9];
  const float* ln_p_b = (const float*)d_in[10];
  const float* Wc     = (const float*)d_in[11];
  const float* bc     = (const float*)d_in[12];
  const float* bfuse  = (const float*)d_in[13];
  const float* Wq     = (const float*)d_in[14];
  const float* Wk     = (const float*)d_in[15];
  const float* Wv     = (const float*)d_in[16];
  const float* Wg     = (const float*)d_in[17];
  const float* bg     = (const float*)d_in[18];
  const float* Wo     = (const float*)d_in[19];
  const float* bo     = (const float*)d_in[20];
  float* out = (float*)d_out;

  // Workspace carve-up (~97.5 MB of f32).
  float* ws   = (float*)d_ws;
  float* ni_ln = ws; ws += (size_t)IDIM * CND;
  float* nj_ln = ws; ws += (size_t)JDIM * CND;
  float* q     = ws; ws += (size_t)IDIM * HCD;
  float* kbuf  = ws; ws += (size_t)JDIM * HCD;
  float* vT    = ws; ws += (size_t)HCD * JDIM;
  float* g     = ws; ws += (size_t)IDIM * HCD;
  float* o     = ws; ws += (size_t)IDIM * HCD;
  float* pb    = ws; ws += (size_t)NH * JDIM * IDIM;

  ln_node_kernel<<<IDIM, 256, 0, stream>>>(node_i, ln_i_a, ln_i_b, ni_ln);
  ln_node_kernel<<<JDIM, 256, 0, stream>>>(node_j, ln_j_a, ln_j_b, nj_ln);
  proj_qkvg_kernel<<<768, 256, 0, stream>>>(ni_ln, nj_ln, Wq, Wk, Wv, Wg, bg,
                                            q, kbuf, vT, g);
  pair_bias_kernel<<<18432, 256, 0, stream>>>(pe, pm, ln_p_a, ln_p_b, Wc, bc, bfuse, pb);
  attention_kernel<<<384, 32 * ATT_WAVES, 0, stream>>>(q, kbuf, vT, pb, g, o);
  out_proj_kernel<<<192, 256, 0, stream>>>(o, Wo, bo, out);
}
